// SelectMolAttachment_76012331205060
// MI455X (gfx1250) — matrix operand order repaired and checked by reading, not compiled, not run on hardware
//
#include <hip/hip_runtime.h>
#include <math.h>

// ---------------------------------------------------------------------------
// CDNA5 / gfx1250 SelectMolAttachment MLP
//   x = [node_hiddens | gather(mol_a_reprs, batch_indices)]  (384)
//   384 ->256 relu ->128 relu ->64 relu ->1 sigmoid
// bf16 WMMA (v_wmma_f32_16x16x32_bf16), fp32 accumulation.
// Per block: 32 rows, 2 waves. Waves split the N (column) range so every
// B-tile (weights) fetched from global is reused by 2 A-tiles (M-blocking):
// halves L2 weight traffic vs per-wave row striping.
// LDS row pitches padded by 8 halfs (4 dwords) -> conflict-free ds_load_b128.
// ---------------------------------------------------------------------------

typedef __attribute__((ext_vector_type(16))) __bf16 v16bf;
typedef __attribute__((ext_vector_type(8)))  __bf16 v8bf;
typedef __attribute__((ext_vector_type(8)))  float  v8f;

#define ROWS 32          // rows (atoms) per block
#define NTHREADS 64      // 2 waves of 32
#define DIM_NH 128
#define DIM_MA 256
#define K1 384
#define N1 256
#define N2 128
#define N3 64
#define PAD 8            // LDS pitch padding in halfs (4 dwords -> bank spread)
#define PX  (K1 + PAD)   // 392
#define P1  (N1 + PAD)   // 264
#define P2  (N2 + PAD)   // 136
#define P3  (N3 + PAD)   // 72

// -------------------- weight convert + transpose (fp32 -> bf16 [N][K]) -----
__global__ void wt_prep_kernel(const float* __restrict__ src, // [K][N] row major
                               __bf16* __restrict__ dst,      // [N][K]
                               int K, int N, int total) {
  int i = blockIdx.x * blockDim.x + threadIdx.x;
  if (i < total) {
    int k = i / N;
    int n = i - k * N;
    dst[n * K + k] = (__bf16)src[i];
  }
}

// -------------------- one MLP layer, both 16-row tiles, half the columns ---
// in  : LDS, ROWS x K bf16, row pitch PIN
// out : LDS, ROWS x N bf16, row pitch POUT
// Wt  : global, N x K bf16 (pre-transposed, unpadded pitch K)
// Wave handles columns [nBase, nBase + N/2). Both M-tiles (rows 0-15, 16-31)
// are computed against each fetched B-tile.
// Lane mapping (16-bit operands, wave32):
//   A: lanes 0-15 -> M=lane, K = k0 + 0..15
//      lanes16-31 -> M=lane-16, K = k0 + 16..31
//   B: lanes 0-15 -> N=lane, K=k0+0..15 ; lanes16-31 -> N=lane-16, K=k0+16..31
//   C/D: lanes 0-15 -> N=lane, M=vgpr ; lanes16-31 -> N=lane-16, M=8+vgpr
template <int K, int N, int PIN, int POUT, int NCHUNK>
__device__ __forceinline__ void mlp_layer(const __bf16* __restrict__ in,
                                          __bf16* __restrict__ out,
                                          const __bf16* __restrict__ Wt,
                                          const float* __restrict__ bias,
                                          int nBase, int lane) {
  const int hi  = lane >> 4;       // lane-half selector
  const int l15 = lane & 15;
  const __bf16* arow0 = in + (l15)      * PIN + hi * 16; // M-tile 0: rows 0-15
  const __bf16* arow1 = in + (16 + l15) * PIN + hi * 16; // M-tile 1: rows 16-31

  for (int n0 = nBase; n0 < nBase + N / 2; n0 += 16 * NCHUNK) {
    v8f acc[2][NCHUNK];
#pragma unroll
    for (int mt = 0; mt < 2; ++mt)
#pragma unroll
      for (int c = 0; c < NCHUNK; ++c)
        acc[mt][c] = v8f{0.f, 0.f, 0.f, 0.f, 0.f, 0.f, 0.f, 0.f};

    for (int k0 = 0; k0 < K; k0 += 32) {
      // A operands: two 16-byte chunks per lane per M-tile
      v8bf a0lo = *(const v8bf*)(arow0 + k0);
      v8bf a0hi = *(const v8bf*)(arow0 + k0 + 8);
      v8bf a1lo = *(const v8bf*)(arow1 + k0);
      v8bf a1hi = *(const v8bf*)(arow1 + k0 + 8);
      v16bf A0 = __builtin_shufflevector(a0lo, a0hi,
                   0,1,2,3,4,5,6,7,8,9,10,11,12,13,14,15);
      v16bf A1 = __builtin_shufflevector(a1lo, a1hi,
                   0,1,2,3,4,5,6,7,8,9,10,11,12,13,14,15);

      const __bf16* bbase = Wt + (size_t)(n0 + l15) * K + k0 + hi * 16;
#pragma unroll
      for (int c = 0; c < NCHUNK; ++c) {
        v16bf B = *(const v16bf*)(bbase + c * 16 * K);
        acc[0][c] = __builtin_amdgcn_wmma_f32_16x16x32_bf16(
            false, A0, false, B, (short)0, acc[0][c], false, false);
        acc[1][c] = __builtin_amdgcn_wmma_f32_16x16x32_bf16(
            false, A1, false, B, (short)0, acc[1][c], false, false);
      }
    }

    // bias + ReLU + store bf16 activations to LDS
#pragma unroll
    for (int c = 0; c < NCHUNK; ++c) {
      int n = n0 + c * 16 + l15;
      float bv = bias[n];
#pragma unroll
      for (int mt = 0; mt < 2; ++mt) {
#pragma unroll
        for (int j = 0; j < 8; ++j) {
          float v = acc[mt][c][j] + bv;
          v = fmaxf(v, 0.0f);
          int m = mt * 16 + hi * 8 + j;
          out[m * POUT + n] = (__bf16)v;
        }
      }
    }
  }
}

// -------------------- main fused kernel ------------------------------------
__global__ __launch_bounds__(NTHREADS)
void mol_attach_mlp_kernel(const float* __restrict__ molA,     // [B, 256]
                           const float* __restrict__ nodeH,    // [BA, 128]
                           const int*   __restrict__ bidx,     // [BA]
                           const __bf16* __restrict__ Wt1, const float* __restrict__ b1,
                           const __bf16* __restrict__ Wt2, const float* __restrict__ b2,
                           const __bf16* __restrict__ Wt3, const float* __restrict__ b3,
                           const __bf16* __restrict__ Wt4, const float* __restrict__ b4,
                           float* __restrict__ out) {
  __shared__ __bf16 sX [ROWS * PX];   // concat input, bf16, padded pitch
  __shared__ __bf16 sH1[ROWS * P1];
  __shared__ __bf16 sH2[ROWS * P2];
  __shared__ __bf16 sH3[ROWS * P3];
  __shared__ int    sIdx[ROWS];

  const int tid = threadIdx.x;
  const int r0  = blockIdx.x * ROWS;

  if (tid < ROWS) sIdx[tid] = bidx[r0 + tid];
  __syncthreads();

  // ---- build X = [node_hiddens | gather(molA)] in bf16 ----
  for (int i = tid; i < ROWS * (DIM_NH / 4); i += NTHREADS) {
    int row = i / (DIM_NH / 4);
    int c4  = i - row * (DIM_NH / 4);
    float4 v = ((const float4*)(nodeH + (size_t)(r0 + row) * DIM_NH))[c4];
    __bf16* d = &sX[row * PX + c4 * 4];
    d[0] = (__bf16)v.x; d[1] = (__bf16)v.y; d[2] = (__bf16)v.z; d[3] = (__bf16)v.w;
  }
  for (int i = tid; i < ROWS * (DIM_MA / 4); i += NTHREADS) {
    int row = i / (DIM_MA / 4);
    int c4  = i - row * (DIM_MA / 4);
    float4 v = ((const float4*)(molA + (size_t)sIdx[row] * DIM_MA))[c4];
    __bf16* d = &sX[row * PX + DIM_NH + c4 * 4];
    d[0] = (__bf16)v.x; d[1] = (__bf16)v.y; d[2] = (__bf16)v.z; d[3] = (__bf16)v.w;
  }
  __syncthreads();

  const int wave = tid >> 5;
  const int lane = tid & 31;

  // Each wave computes ALL 32 rows over its half of the output columns:
  // every fetched B-tile feeds 2 WMMAs (M-tiles 0 and 1).
  mlp_layer<K1, N1, PX, P1, 4>(sX,  sH1, Wt1, b1, wave * (N1 / 2), lane);
  __syncthreads();
  mlp_layer<N1, N2, P1, P2, 4>(sH1, sH2, Wt2, b2, wave * (N2 / 2), lane);
  __syncthreads();
  mlp_layer<N2, N3, P2, P3, 2>(sH2, sH3, Wt3, b3, wave * (N3 / 2), lane);
  __syncthreads();

  // ---- layer 4: 64 -> 1, sigmoid; one row per thread ----
  if (tid < ROWS) {
    float acc = b4[0];
#pragma unroll
    for (int k = 0; k < N3; ++k)
      acc += (float)sH3[tid * P3 + k] * (float)Wt4[k];
    out[r0 + tid] = 1.0f / (1.0f + __expf(-acc));
  }
}

// -------------------- launch ------------------------------------------------
extern "C" void kernel_launch(void* const* d_in, const int* in_sizes, int n_in,
                              void* d_out, int out_size, void* d_ws, size_t ws_size,
                              hipStream_t stream) {
  const float* molA  = (const float*)d_in[0];
  const float* nodeH = (const float*)d_in[1];
  const int*   bidx  = (const int*)d_in[2];
  const float* W1 = (const float*)d_in[3];  const float* b1 = (const float*)d_in[4];
  const float* W2 = (const float*)d_in[5];  const float* b2 = (const float*)d_in[6];
  const float* W3 = (const float*)d_in[7];  const float* b3 = (const float*)d_in[8];
  const float* W4 = (const float*)d_in[9];  const float* b4 = (const float*)d_in[10];
  float* out = (float*)d_out;

  // bf16 transposed weights in workspace
  __bf16* ws  = (__bf16*)d_ws;
  __bf16* Wt1 = ws;                       // 256 x 384
  __bf16* Wt2 = ws + 384 * 256;           // 128 x 256
  __bf16* Wt3 = Wt2 + 256 * 128;          //  64 x 128
  __bf16* Wt4 = Wt3 + 128 * 64;           //  64 (vector)

  wt_prep_kernel<<<(384 * 256 + 255) / 256, 256, 0, stream>>>(W1, Wt1, 384, 256, 384 * 256);
  wt_prep_kernel<<<(256 * 128 + 255) / 256, 256, 0, stream>>>(W2, Wt2, 256, 128, 256 * 128);
  wt_prep_kernel<<<(128 * 64  + 255) / 256, 256, 0, stream>>>(W3, Wt3, 128, 64, 128 * 64);
  wt_prep_kernel<<<(64        + 255) / 256, 256, 0, stream>>>(W4, Wt4, 64, 1, 64);

  const int BA = in_sizes[2];             // number of B atoms
  mol_attach_mlp_kernel<<<BA / ROWS, NTHREADS, 0, stream>>>(
      molA, nodeH, bidx, Wt1, b1, Wt2, b2, Wt3, b3, Wt4, b4, out);
}
